// LSTMEncoder_9887014715517
// MI455X (gfx1250) — matrix operand order repaired and checked by reading, not compiled, not run on hardware
//
#include <hip/hip_runtime.h>
#include <hip/hip_bf16.h>

typedef __attribute__((ext_vector_type(16))) __bf16 v16bf;
typedef __attribute__((ext_vector_type(8)))  __bf16 v8bf;
typedef __attribute__((ext_vector_type(8)))  float  v8f;

#define NB 64
#define NT 512
#define NH 1024
#define N4H 4096

__device__ __forceinline__ float sigf(float x) { return 1.0f / (1.0f + __expf(-x)); }

union AU { v16bf v; v8bf h[2]; };

// A-fragment (16-bit A, 16x32, row-major source):
// lane L: row = L%16, kbase = (L/16)*16; elems 0..15 -> K=kbase+0..15
static __device__ __forceinline__ v16bf load_a(const __bf16* A, int lda, int row0, int k0, int lane) {
  const int row = row0 + (lane & 15);
  const int kb  = k0 + ((lane >> 4) << 4);
  AU u;
  u.h[0] = *(const v8bf*)(A + (size_t)row * lda + kb);
  u.h[1] = *(const v8bf*)(A + (size_t)row * lda + kb + 8);
  return u.v;
}

// B-fragment from pre-packed weights: fragment f = nt*kTiles + kt, lane-major 16 bf16 per lane.
static __device__ __forceinline__ v16bf load_b(const __bf16* P, int kTiles, int nt, int kt, int lane) {
  return *(const v16bf*)(P + ((size_t)nt * kTiles + kt) * 512 + lane * 16);
}

static __device__ __forceinline__ v8f wmma_bf16(v16bf a, v16bf b, v8f c) {
  return __builtin_amdgcn_wmma_f32_16x16x32_bf16(false, a, false, b, (short)0, c, false, false);
}

__global__ void zero_ws_kernel(unsigned int* p, long n) {
  long i = (long)blockIdx.x * blockDim.x + threadIdx.x;
  const long stride = (long)gridDim.x * blockDim.x;
  for (; i < n; i += stride) p[i] = 0u;
}

__global__ void cvt_bf16_kernel(const float* __restrict__ x, __bf16* __restrict__ y, long n) {
  long i = (long)blockIdx.x * blockDim.x + threadIdx.x;
  const long stride = (long)gridDim.x * blockDim.x;
  for (; i < n; i += stride) y[i] = (__bf16)x[i];
}

// Pack W[krow0 + k, n] (row-major [*, 4096] fp32) into per-lane WMMA B layout (bf16).
// B order used here (contiguous k, same as load_a): lane L holds column n = L%16; K = (L/16)*16 + e, e=0..15.
__global__ void pack_w_kernel(const float* __restrict__ W, __bf16* __restrict__ P,
                              int kTiles, int krow0, long total) {
  long i = (long)blockIdx.x * blockDim.x + threadIdx.x;
  const long stride = (long)gridDim.x * blockDim.x;
  for (; i < total; i += stride) {
    const long f = i >> 9;
    const int  r = (int)(i & 511);
    const int lane = r >> 4;
    const int e = r & 15;
    const int nt = (int)(f / kTiles);
    const int kt = (int)(f - (long)nt * kTiles);
    const int n = nt * 16 + (lane & 15);
    const int k = kt * 32 + ((lane >> 4) << 4) + e;
    P[i] = (__bf16)W[(size_t)(krow0 + k) * N4H + n];
  }
}

// Zx[m,n] = sum_k Xb[m,k] * W0x[k,n] + b0[n];  M=32768, N=4096, K=1024.
// Wave-task: one 16-row M tile x four 16-col N tiles; double-buffered fragments.
__global__ __launch_bounds__(256) void gemm_zx_kernel(const __bf16* __restrict__ Xb,
                                                      const __bf16* __restrict__ Wp,
                                                      const float* __restrict__ bs,
                                                      float* __restrict__ Zx) {
  const int lane = threadIdx.x & 31;
  const int task = blockIdx.x * 8 + (threadIdx.x >> 5); // 131072 wave-tasks
  const int mt = task >> 6;                             // 0..2047 (64 waves = 8 blocks share mt)
  const int ng = task & 63;                             // group of 4 n-tiles
  v8f acc[4] = {};
  v16bf a = load_a(Xb, NH, mt * 16, 0, lane);
  v16bf b[4];
#pragma unroll
  for (int g = 0; g < 4; ++g) b[g] = load_b(Wp, 32, ng * 4 + g, 0, lane);
#pragma unroll 2
  for (int kt = 0; kt < 32; ++kt) {
    const int kn = (kt + 1 < 32) ? kt + 1 : kt;
    v16bf an = load_a(Xb, NH, mt * 16, kn * 32, lane);
    v16bf bn[4];
#pragma unroll
    for (int g = 0; g < 4; ++g) bn[g] = load_b(Wp, 32, ng * 4 + g, kn, lane);
#pragma unroll
    for (int g = 0; g < 4; ++g) acc[g] = wmma_bf16(a, b[g], acc[g]);
    a = an;
#pragma unroll
    for (int g = 0; g < 4; ++g) b[g] = bn[g];
  }
  const int rh = (lane >> 4) << 3;
  const int cl = lane & 15;
#pragma unroll
  for (int g = 0; g < 4; ++g) {
    const int col = (ng * 4 + g) * 16 + cl;
    const float bv = bs[col];
#pragma unroll
    for (int r = 0; r < 8; ++r) {
      const int m = mt * 16 + rh + r;
      Zx[(size_t)m * N4H + col] = acc[g][r] + bv;
    }
  }
}

// Layer-0 step: z = h_prev @ Wh0 + Zx[:, t, :]; fused LSTM cell + masking.
// Wave-task: mPair p (2 M tiles) x hTile j; gates i/j/f/o are n-tiles g*64+j.
__global__ __launch_bounds__(128) void lstm_step0_kernel(
    const __bf16* __restrict__ hIn, __bf16* __restrict__ hOut,
    const __bf16* __restrict__ Whp, const float* __restrict__ Zx,
    const int* __restrict__ lens,
    float* __restrict__ hF, float* __restrict__ cF,
    __bf16* __restrict__ aCur, int t) {
  const int lane = threadIdx.x & 31;
  const int task = blockIdx.x * 4 + (threadIdx.x >> 5); // 0..127
  const int p = task & 1;
  const int j = task >> 1;
  v8f acc[2][4] = {};
  v16bf a0 = load_a(hIn, NH, p * 32, 0, lane);
  v16bf a1 = load_a(hIn, NH, p * 32 + 16, 0, lane);
  v16bf b[4];
#pragma unroll
  for (int g = 0; g < 4; ++g) b[g] = load_b(Whp, 32, g * 64 + j, 0, lane);
#pragma unroll 2
  for (int kt = 0; kt < 32; ++kt) {
    const int kn = (kt + 1 < 32) ? kt + 1 : kt;
    v16bf an0 = load_a(hIn, NH, p * 32, kn * 32, lane);
    v16bf an1 = load_a(hIn, NH, p * 32 + 16, kn * 32, lane);
    v16bf bn[4];
#pragma unroll
    for (int g = 0; g < 4; ++g) bn[g] = load_b(Whp, 32, g * 64 + j, kn, lane);
#pragma unroll
    for (int g = 0; g < 4; ++g) {
      acc[0][g] = wmma_bf16(a0, b[g], acc[0][g]);
      acc[1][g] = wmma_bf16(a1, b[g], acc[1][g]);
    }
    a0 = an0; a1 = an1;
#pragma unroll
    for (int g = 0; g < 4; ++g) b[g] = bn[g];
  }
  const int rh = (lane >> 4) << 3;
  const int h = j * 16 + (lane & 15);
#pragma unroll
  for (int mm = 0; mm < 2; ++mm) {
#pragma unroll
    for (int r = 0; r < 8; ++r) {
      const int bb = p * 32 + mm * 16 + rh + r;
      const size_t zb = ((size_t)bb * NT + t) * N4H;
      const float zi = acc[mm][0][r] + Zx[zb + h];
      const float zj = acc[mm][1][r] + Zx[zb + NH + h];
      const float zf = acc[mm][2][r] + Zx[zb + 2 * NH + h];
      const float zo = acc[mm][3][r] + Zx[zb + 3 * NH + h];
      const size_t si = (size_t)bb * NH + h;
      const float cp = cF[si];
      const float cn = cp * sigf(zf + 1.0f) + sigf(zi) * tanhf(zj);
      const float hn = tanhf(cn) * sigf(zo);
      const bool m = t < lens[bb];
      const float ck = m ? cn : cp;
      const float hk = m ? hn : hF[si];
      cF[si] = ck;
      hF[si] = hk;
      hOut[si] = (__bf16)hk;
      aCur[(size_t)bb * 2048 + h] = (__bf16)(m ? hn : 0.0f); // y0 -> layer-1 input
    }
  }
}

// Layer-1 step: z = [y0_t | h1_prev] @ W1 + b1 (K=2048); fused cell; writes y1 to d_out.
__global__ __launch_bounds__(128) void lstm_step1_kernel(
    const __bf16* __restrict__ aIn, __bf16* __restrict__ aNext,
    const __bf16* __restrict__ Wp, const float* __restrict__ bs,
    const int* __restrict__ lens,
    float* __restrict__ hF, float* __restrict__ cF,
    float* __restrict__ y, int t) {
  const int lane = threadIdx.x & 31;
  const int task = blockIdx.x * 4 + (threadIdx.x >> 5);
  const int p = task & 1;
  const int j = task >> 1;
  v8f acc[2][4] = {};
  v16bf a0 = load_a(aIn, 2048, p * 32, 0, lane);
  v16bf a1 = load_a(aIn, 2048, p * 32 + 16, 0, lane);
  v16bf b[4];
#pragma unroll
  for (int g = 0; g < 4; ++g) b[g] = load_b(Wp, 64, g * 64 + j, 0, lane);
#pragma unroll 2
  for (int kt = 0; kt < 64; ++kt) {
    const int kn = (kt + 1 < 64) ? kt + 1 : kt;
    v16bf an0 = load_a(aIn, 2048, p * 32, kn * 32, lane);
    v16bf an1 = load_a(aIn, 2048, p * 32 + 16, kn * 32, lane);
    v16bf bn[4];
#pragma unroll
    for (int g = 0; g < 4; ++g) bn[g] = load_b(Wp, 64, g * 64 + j, kn, lane);
#pragma unroll
    for (int g = 0; g < 4; ++g) {
      acc[0][g] = wmma_bf16(a0, b[g], acc[0][g]);
      acc[1][g] = wmma_bf16(a1, b[g], acc[1][g]);
    }
    a0 = an0; a1 = an1;
#pragma unroll
    for (int g = 0; g < 4; ++g) b[g] = bn[g];
  }
  const int rh = (lane >> 4) << 3;
  const int h = j * 16 + (lane & 15);
  const float bi = bs[h];
  const float bj = bs[NH + h];
  const float bf = bs[2 * NH + h];
  const float bo = bs[3 * NH + h];
#pragma unroll
  for (int mm = 0; mm < 2; ++mm) {
#pragma unroll
    for (int r = 0; r < 8; ++r) {
      const int bb = p * 32 + mm * 16 + rh + r;
      const size_t si = (size_t)bb * NH + h;
      const float cp = cF[si];
      const float cn = cp * sigf(acc[mm][2][r] + bf + 1.0f) +
                       sigf(acc[mm][0][r] + bi) * tanhf(acc[mm][1][r] + bj);
      const float hn = tanhf(cn) * sigf(acc[mm][3][r] + bo);
      const bool m = t < lens[bb];
      const float ck = m ? cn : cp;
      const float hk = m ? hn : hF[si];
      cF[si] = ck;
      hF[si] = hk;
      aNext[(size_t)bb * 2048 + NH + h] = (__bf16)hk; // h1 for next step's A
      y[((size_t)bb * NT + t) * NH + h] = m ? hn : 0.0f;
    }
  }
}

__global__ void finalize_kernel(const float* __restrict__ h0, const float* __restrict__ h1,
                                const float* __restrict__ c0, const float* __restrict__ c1,
                                float* __restrict__ out) {
  const int i = blockIdx.x * blockDim.x + threadIdx.x;
  if (i < NB * NH) {
    const size_t base = (size_t)NB * NT * NH;
    out[base + i] = h0[i];
    out[base + NB * NH + i] = h1[i];
    out[base + 2 * NB * NH + i] = c0[i];
    out[base + 3 * NB * NH + i] = c1[i];
  }
}

extern "C" void kernel_launch(void* const* d_in, const int* in_sizes, int n_in,
                              void* d_out, int out_size, void* d_ws, size_t ws_size,
                              hipStream_t stream) {
  (void)in_sizes; (void)n_in; (void)out_size; (void)ws_size;
  const float* x    = (const float*)d_in[0];
  const int*   lens = (const int*)d_in[1];
  const float* W0   = (const float*)d_in[2];
  const float* b0   = (const float*)d_in[3];
  const float* W1   = (const float*)d_in[4];
  const float* b1   = (const float*)d_in[5];
  float* out = (float*)d_out;

  char* ws = (char*)d_ws;
  size_t off = 0;
  auto take = [&](size_t bytes) -> char* {
    char* p = ws + off;
    off += (bytes + 255) & ~(size_t)255;
    return p;
  };

  __bf16* xb   = (__bf16*)take((size_t)NB * NT * NH * 2);  // 64 MB
  __bf16* w0xp = (__bf16*)take((size_t)NH * N4H * 2);      // 8 MB
  __bf16* w0hp = (__bf16*)take((size_t)NH * N4H * 2);      // 8 MB
  __bf16* w1p  = (__bf16*)take((size_t)2 * NH * N4H * 2);  // 16 MB
  float*  Zx   = (float*)take((size_t)NB * NT * N4H * 4);  // 512 MB
  char* zbase = ws + off;                                   // zero-init region start
  float*  h0F  = (float*)take((size_t)NB * NH * 4);
  float*  c0F  = (float*)take((size_t)NB * NH * 4);
  float*  h1F  = (float*)take((size_t)NB * NH * 4);
  float*  c1F  = (float*)take((size_t)NB * NH * 4);
  __bf16* h0b0 = (__bf16*)take((size_t)NB * NH * 2);
  __bf16* h0b1 = (__bf16*)take((size_t)NB * NH * 2);
  __bf16* a1b0 = (__bf16*)take((size_t)NB * 2048 * 2);
  __bf16* a1b1 = (__bf16*)take((size_t)NB * 2048 * 2);
  const long zwords = (long)(((ws + off) - zbase) / 4);

  zero_ws_kernel<<<512, 256, 0, stream>>>((unsigned int*)zbase, zwords);
  cvt_bf16_kernel<<<2048, 256, 0, stream>>>(x, xb, (long)NB * NT * NH);
  pack_w_kernel<<<2048, 256, 0, stream>>>(W0, w0xp, 32, 0, (long)NH * N4H);
  pack_w_kernel<<<2048, 256, 0, stream>>>(W0, w0hp, 32, NH, (long)NH * N4H);
  pack_w_kernel<<<2048, 256, 0, stream>>>(W1, w1p, 64, 0, (long)2 * NH * N4H);
  gemm_zx_kernel<<<16384, 256, 0, stream>>>(xb, w0xp, b0, Zx);

  for (int t = 0; t < NT; ++t) {
    __bf16* hIn   = (t & 1) ? h0b1 : h0b0;
    __bf16* hOut  = (t & 1) ? h0b0 : h0b1;
    __bf16* aCur  = (t & 1) ? a1b1 : a1b0;
    __bf16* aNext = (t & 1) ? a1b0 : a1b1;
    lstm_step0_kernel<<<32, 128, 0, stream>>>(hIn, hOut, w0hp, Zx, lens, h0F, c0F, aCur, t);
    lstm_step1_kernel<<<32, 128, 0, stream>>>(aCur, aNext, w1p, b1, lens, h1F, c1F, out, t);
  }
  finalize_kernel<<<(NB * NH + 255) / 256, 256, 0, stream>>>(h0F, h1F, c0F, c1F, out);
}
